// DRNet_Retinex_time_36988258353244
// MI455X (gfx1250) — matrix operand order repaired and checked by reading, not compiled, not run on hardware
//
#include <hip/hip_runtime.h>

// ---------------------------------------------------------------------------
// MI455X (gfx1250) implementation.
//  * All convs with Cin>=96 run as implicit-GEMM on v_wmma_f32_16x16x32_bf16
//    (bf16 A/B, fp32 accumulate) -- the highest-throughput dense CDNA5 path
//    with fp32 accumulation. LN / softmax / norms / small convs stay fp32.
//  * GEMM is software-pipelined: register prefetch + LDS ping-pong, one
//    barrier per K-step; im2col decode is incremental (no divides in loop).
//  * Workspace requirement: ~212 MB (static offset map with buffer reuse).
// ---------------------------------------------------------------------------

typedef __attribute__((ext_vector_type(16))) __bf16 v16bf;
typedef __attribute__((ext_vector_type(8)))  float  v8f;

#define BN      4
#define CDIM    96
#define NHEADS  4
#define CHD     24
#define HDIM    128
#define WDIM    128
#define SDIM    (HDIM * WDIM)        // 16384

__device__ __forceinline__ float geluf(float x) {
    return 0.5f * x * (1.0f + erff(x * 0.70710678118654752440f));
}

// -------------------------------- weight pack ------------------------------
// conv weight [Cout][CinReal][kh][kw] f32 -> bf16 [MP][KP], K = ci*khw + r,
// zero padding for m>=Cout or ci>=CinReal.
__global__ void pack_weight_k(const float* __restrict__ w, __bf16* __restrict__ dst,
                              int Cout, int CinReal, int khw, int MP, int KP) {
    int i = blockIdx.x * blockDim.x + threadIdx.x;
    if (i >= MP * KP) return;
    int m = i / KP, k = i - m * KP;
    int ci = k / khw, r = k - ci * khw;
    float v = 0.0f;
    if (m < Cout && ci < CinReal) v = w[((size_t)m * CinReal + ci) * khw + r];
    dst[i] = (__bf16)v;
}

__global__ void f32_to_bf16_k(const float* __restrict__ in, __bf16* __restrict__ out, int n) {
    int i = blockIdx.x * blockDim.x + threadIdx.x;
    if (i < n) out[i] = (__bf16)in[i];
}

// ------------------------- implicit-GEMM conv (WMMA) -----------------------
// D[m, pix] = sum_k W[m,k] * im2col(X)[k, pix];  k = ci*KHW + (ky*3+kx)
// Block: 256 threads (8 wave32). Tile M=32 x N=128 (one full image row).
// Ping-pong LDS + register prefetch: 1 barrier / K-step, loads overlap WMMA.
#define LDSP 40   // LDS row pitch in halves (80B: 16B-aligned, conflict-spread)

template <int KHW>
__global__ __launch_bounds__(256)
void gemm_conv_bf16(const __bf16* __restrict__ W, const __bf16* __restrict__ X,
                    const float* __restrict__ bias,
                    const float* __restrict__ res1, const float* __restrict__ res2,
                    float* __restrict__ outF, __bf16* __restrict__ outB,
                    int M, int K, int Cin, int doGelu) {
    __shared__ __align__(16) __bf16 As[2][32 * LDSP];    //  2 x 2.5 KB
    __shared__ __align__(16) __bf16 Bs[2][128 * LDSP];   //  2 x 10 KB

    const int tid   = threadIdx.x;
    const int mBase = blockIdx.y * 32;
    const int row   = blockIdx.x;        // 0 .. BN*HDIM-1
    const int n     = row >> 7;
    const int y0    = row & (HDIM - 1);
    const int p0    = y0 * WDIM;

    const int lane    = tid & 31;
    const int wv      = tid >> 5;        // 8 waves
    const int msub    = (wv & 1) * 16;
    const int nsub    = (wv >> 1) * 32;  // two 16-col tiles per wave
    const int halfSel = (lane >> 4) & 1;
    const int mrow    = lane & 15;

    // B staging ownership: kl = K-slot, subn = pixel-phase (interleaved)
    const int kl   = tid >> 3;           // 0..31
    const int subn = tid & 7;            // pixels subn, subn+8, ... (16 of them)

    // incremental im2col decode state for k = k0 + kl
    int ci, rr;
    if (KHW == 9) { ci = kl / 9; rr = kl - ci * 9; }
    else          { ci = kl;     rr = 0; }

    v8f acc0 = {}; v8f acc1 = {};

    uint4  aReg = {0, 0, 0, 0};
    __bf16 bReg[16];

    // ---- prefetch tile k0 = 0 ----
    if (tid < 128)
        aReg = *(const uint4*)(W + (size_t)(mBase + (tid >> 2)) * K + (tid & 3) * 8);
    {
        int dy = 0, dx = 0;
        if (KHW == 9) { int ry = (rr * 11) >> 5; dy = ry - 1; dx = rr - ry * 3 - 1; }
        int yy = y0 + dy;
        bool yok = (unsigned)yy < (unsigned)HDIM;
        const __bf16* src = X + (((size_t)n * Cin + ci) * HDIM + yy) * WDIM;
        #pragma unroll
        for (int j = 0; j < 16; ++j) {
            int nl = subn + 8 * j;
            if (KHW == 1) { bReg[j] = src[nl]; }
            else {
                int xx = nl + dx;
                __bf16 v = (__bf16)0.0f;
                if (yok && (unsigned)xx < (unsigned)WDIM) v = src[xx];
                bReg[j] = v;
            }
        }
    }

    int p = 0;
    for (int k0 = 0; k0 < K; k0 += 32) {
        // ---- commit prefetched registers into LDS buffer p ----
        if (tid < 128)
            *(uint4*)(&As[p][(tid >> 2) * LDSP + (tid & 3) * 8]) = aReg;
        #pragma unroll
        for (int j = 0; j < 16; ++j)
            Bs[p][(subn + 8 * j) * LDSP + kl] = bReg[j];
        __syncthreads();

        // ---- prefetch tile k0+32 (overlaps the WMMAs below) ----
        if (k0 + 32 < K) {
            int kn = k0 + 32;
            if (tid < 128) {
                aReg = *(const uint4*)(W + (size_t)(mBase + (tid >> 2)) * K + kn + (tid & 3) * 8);
                if (kn + 32 < K)   // prefetch weight row two steps ahead into cache
                    __builtin_prefetch((const void*)(W + (size_t)(mBase + (tid >> 2)) * K +
                                                     kn + 32 + (tid & 3) * 8), 0, 0);
            }
            if (KHW == 9) { ci += 3; rr += 5; if (rr >= 9) { rr -= 9; ++ci; } }
            else          { ci += 32; }
            int dy = 0, dx = 0;
            if (KHW == 9) { int ry = (rr * 11) >> 5; dy = ry - 1; dx = rr - ry * 3 - 1; }
            int yy = y0 + dy;
            bool yok = (unsigned)yy < (unsigned)HDIM;
            const __bf16* src = X + (((size_t)n * Cin + ci) * HDIM + yy) * WDIM;
            #pragma unroll
            for (int j = 0; j < 16; ++j) {
                int nl = subn + 8 * j;
                if (KHW == 1) { bReg[j] = src[nl]; }
                else {
                    int xx = nl + dx;
                    __bf16 v = (__bf16)0.0f;
                    if (yok && (unsigned)xx < (unsigned)WDIM) v = src[xx];
                    bReg[j] = v;
                }
            }
        }

        // ---- fragments per ISA VGPR layouts (16B ds_read runs) ----
        union { uint4 u[2]; v16bf v; } af, bf0, bf1;
        {
            const __bf16* ar = &As[p][(msub + mrow) * LDSP];
            int r0 = halfSel * 8;
            af.u[0] = *(const uint4*)(ar + r0);
            af.u[1] = *(const uint4*)(ar + 16 + r0);
        }
        {
            int krun = halfSel * 8;
            const __bf16* b0 = &Bs[p][(nsub + mrow) * LDSP + krun];
            bf0.u[0] = *(const uint4*)(b0);
            bf0.u[1] = *(const uint4*)(b0 + 16);
            const __bf16* b1 = &Bs[p][(nsub + 16 + mrow) * LDSP + krun];
            bf1.u[0] = *(const uint4*)(b1);
            bf1.u[1] = *(const uint4*)(b1 + 16);
        }
        acc0 = __builtin_amdgcn_wmma_f32_16x16x32_bf16(false, af.v, false, bf0.v,
                                                       (short)0, acc0, false, false);
        acc1 = __builtin_amdgcn_wmma_f32_16x16x32_bf16(false, af.v, false, bf1.v,
                                                       (short)0, acc1, false, false);
        p ^= 1;
    }

    // ---- epilogue: bias / gelu / residuals / dual-precision store ----
    #pragma unroll
    for (int j = 0; j < 2; ++j) {
        v8f a = j ? acc1 : acc0;
        #pragma unroll
        for (int g = 0; g < 8; ++g) {
            int m   = mBase + msub + g + halfSel * 8;
            int pix = p0 + nsub + j * 16 + mrow;
            float v = a[g];
            if (bias)   v += bias[m];
            if (doGelu) v  = geluf(v);
            size_t o = ((size_t)n * M + m) * SDIM + pix;
            if (res1) v += res1[o];
            if (res2) v += res2[o];
            if (outF) outF[o] = v;
            if (outB) outB[o] = (__bf16)v;
        }
    }
}

// ------------------------------ small convs --------------------------------
__global__ void conv1x1_f32_k(const float* __restrict__ in, const float* __restrict__ w,
                              const float* __restrict__ b, float* __restrict__ out,
                              int Cin, int Cout, int total) {
    int idx = blockIdx.x * blockDim.x + threadIdx.x;
    if (idx >= total) return;
    int pix = idx % SDIM; int t = idx / SDIM;
    int co  = t % Cout;   int n = t / Cout;
    float s = b ? b[co] : 0.0f;
    const float* base = in + (size_t)n * Cin * SDIM + pix;
    for (int ci = 0; ci < Cin; ++ci) s += base[(size_t)ci * SDIM] * w[co * Cin + ci];
    out[((size_t)n * Cout + co) * SDIM + pix] = s;
}

__global__ void conv3x3_direct_k(const float* __restrict__ in, const float* __restrict__ w,
                                 const float* __restrict__ b, float* __restrict__ out,
                                 int Cin, int Cout, int doGelu, int total) {
    int idx = blockIdx.x * blockDim.x + threadIdx.x;
    if (idx >= total) return;
    int pix = idx % SDIM; int t = idx / SDIM;
    int co  = t % Cout;   int n = t / Cout;
    int y = pix / WDIM, x = pix - y * WDIM;
    float s = b[co];
    for (int ci = 0; ci < Cin; ++ci) {
        const float* src = in + ((size_t)n * Cin + ci) * SDIM;
        const float* wp  = w + ((size_t)co * Cin + ci) * 9;
        #pragma unroll
        for (int ky = 0; ky < 3; ++ky) {
            int yy = y + ky - 1; if (yy < 0 || yy >= HDIM) continue;
            #pragma unroll
            for (int kx = 0; kx < 3; ++kx) {
                int xx = x + kx - 1; if (xx < 0 || xx >= WDIM) continue;
                s += src[yy * WDIM + xx] * wp[ky * 3 + kx];
            }
        }
    }
    if (doGelu) s = geluf(s);
    out[((size_t)n * Cout + co) * SDIM + pix] = s;
}

__global__ void dwconv3x3_k(const __bf16* __restrict__ in, const float* __restrict__ w,
                            float* __restrict__ out, int C, int total) {
    int idx = blockIdx.x * blockDim.x + threadIdx.x;
    if (idx >= total) return;
    int pix = idx % SDIM; int t = idx / SDIM;
    int c   = t % C;      int n = t / C;
    int y = pix / WDIM, x = pix - y * WDIM;
    const __bf16* src = in + ((size_t)n * C + c) * SDIM;
    const float* wp = w + (size_t)c * 9;
    float s = 0.0f;
    #pragma unroll
    for (int ky = 0; ky < 3; ++ky) {
        int yy = y + ky - 1; if (yy < 0 || yy >= HDIM) continue;
        #pragma unroll
        for (int kx = 0; kx < 3; ++kx) {
            int xx = x + kx - 1; if (xx < 0 || xx >= WDIM) continue;
            s += (float)src[yy * WDIM + xx] * wp[ky * 3 + kx];
        }
    }
    out[((size_t)n * C + c) * SDIM + pix] = s;
}

// --------------------------- LayerNorm (channel) ---------------------------
__global__ void layernorm_c_k(const float* __restrict__ x, const float* __restrict__ g,
                              const float* __restrict__ b, __bf16* __restrict__ out,
                              int C, int total) {
    int p = blockIdx.x * blockDim.x + threadIdx.x;
    if (p >= total) return;
    int n = p / SDIM, pix = p - n * SDIM;
    const float* base = x + (size_t)n * C * SDIM + pix;
    float mu = 0.0f;
    for (int c = 0; c < C; ++c) mu += base[(size_t)c * SDIM];
    mu /= (float)C;
    float var = 0.0f;
    for (int c = 0; c < C; ++c) { float d = base[(size_t)c * SDIM] - mu; var += d * d; }
    var /= (float)C;
    float inv = rsqrtf(var + 1e-5f);
    __bf16* ob = out + (size_t)n * C * SDIM + pix;
    for (int c = 0; c < C; ++c)
        ob[(size_t)c * SDIM] = (__bf16)((base[(size_t)c * SDIM] - mu) * inv * g[c] + b[c]);
}

// ------------------------------- attention ---------------------------------
__global__ void rownorm_k(const float* __restrict__ x, float* __restrict__ nrm, int imgC) {
    int bc = blockIdx.x;                // 0..BN*CDIM-1
    int n = bc / CDIM, c = bc - n * CDIM;
    const float* p = x + ((size_t)n * imgC + c) * SDIM;
    __shared__ float red[256];
    float s = 0.0f;
    for (int i = threadIdx.x; i < SDIM; i += 256) { float v = p[i]; s += v * v; }
    red[threadIdx.x] = s; __syncthreads();
    for (int off = 128; off > 0; off >>= 1) {
        if (threadIdx.x < off) red[threadIdx.x] += red[threadIdx.x + off];
        __syncthreads();
    }
    if (threadIdx.x == 0) nrm[bc] = fmaxf(sqrtf(red[0]), 1e-12f);
}

__global__ void gram_k(const float* __restrict__ q, const float* __restrict__ kvf,
                       float* __restrict__ G) {
    int row = blockIdx.x;               // n*96 + h*24 + c
    int h = (row / CHD) % NHEADS; int n = row / (CHD * NHEADS);
    const float* qp = q + (size_t)row * SDIM;
    const float* kb = kvf + ((size_t)n * 2 * CDIM + h * CHD) * SDIM;   // k = ch 0..95
    float acc[CHD];
    #pragma unroll
    for (int d = 0; d < CHD; ++d) acc[d] = 0.0f;
    for (int i = threadIdx.x; i < SDIM; i += 256) {
        float qa = qp[i];
        #pragma unroll
        for (int d = 0; d < CHD; ++d) acc[d] += qa * kb[(size_t)d * SDIM + i];
    }
    __shared__ float red[256];
    for (int d = 0; d < CHD; ++d) {
        red[threadIdx.x] = acc[d]; __syncthreads();
        for (int off = 128; off > 0; off >>= 1) {
            if (threadIdx.x < off) red[threadIdx.x] += red[threadIdx.x + off];
            __syncthreads();
        }
        if (threadIdx.x == 0) G[(size_t)row * CHD + d] = red[0];
        __syncthreads();
    }
}

__global__ void attn_softmax_k(const float* __restrict__ G, const float* __restrict__ nq,
                               const float* __restrict__ nk, const float* __restrict__ temp,
                               float* __restrict__ A) {
    int t = threadIdx.x;                // 0..383 : n*96 + h*24 + c
    int h = (t / CHD) % NHEADS; int n = t / (CHD * NHEADS);
    float qn = nq[t];
    float tm = temp[h];
    float l[CHD]; float mx = -3.4e38f;
    #pragma unroll
    for (int d = 0; d < CHD; ++d) {
        l[d] = G[(size_t)t * CHD + d] / (qn * nk[n * CDIM + h * CHD + d]) * tm;
        mx = fmaxf(mx, l[d]);
    }
    float s = 0.0f;
    #pragma unroll
    for (int d = 0; d < CHD; ++d) { l[d] = expf(l[d] - mx); s += l[d]; }
    float inv = 1.0f / s;
    #pragma unroll
    for (int d = 0; d < CHD; ++d) A[(size_t)t * CHD + d] = l[d] * inv;
}

__global__ void attn_v_k(const float* __restrict__ A, const float* __restrict__ kvf,
                         __bf16* __restrict__ outb) {
    int chunk = blockIdx.x & 63;        // SDIM/256 = 64 chunks
    int bh = blockIdx.x >> 6;
    int h = bh % NHEADS, n = bh / NHEADS;
    __shared__ float sh[CHD * CHD];
    for (int i = threadIdx.x; i < CHD * CHD; i += 256)
        sh[i] = A[(size_t)(n * NHEADS + h) * CHD * CHD + i];
    __syncthreads();
    int pix = chunk * 256 + threadIdx.x;
    float v[CHD];
    #pragma unroll
    for (int d = 0; d < CHD; ++d)
        v[d] = kvf[((size_t)n * 2 * CDIM + CDIM + h * CHD + d) * SDIM + pix];
    #pragma unroll
    for (int c = 0; c < CHD; ++c) {
        float o = 0.0f;
        #pragma unroll
        for (int d = 0; d < CHD; ++d) o += sh[c * CHD + d] * v[d];
        outb[((size_t)n * CDIM + h * CHD + c) * SDIM + pix] = (__bf16)o;
    }
}

// ------------------------------- launcher ----------------------------------
extern "C" void kernel_launch(void* const* d_in, const int* in_sizes, int n_in,
                              void* d_out, int out_size, void* d_ws, size_t ws_size,
                              hipStream_t stream) {
    (void)in_sizes; (void)n_in; (void)out_size; (void)ws_size;

    const float* x     = (const float*)d_in[0];
    const float* Lc_w1 = (const float*)d_in[1];  const float* Lc_b1 = (const float*)d_in[2];
    const float* Lc_w2 = (const float*)d_in[3];  const float* Lc_b2 = (const float*)d_in[4];
    const float* Rc_w1 = (const float*)d_in[5];  const float* Rc_b1 = (const float*)d_in[6];
    const float* Rc_w2 = (const float*)d_in[7];  const float* Rc_b2 = (const float*)d_in[8];
    const float* Lr_w  = (const float*)d_in[9];  const float* Lr_b  = (const float*)d_in[10];
    const float* Rr_w  = (const float*)d_in[11]; const float* Rr_b  = (const float*)d_in[12];
    const float* n1_w  = (const float*)d_in[13]; const float* n1_b  = (const float*)d_in[14];
    const float* nL_w  = (const float*)d_in[15]; const float* nL_b  = (const float*)d_in[16];
    const float* n2_w  = (const float*)d_in[17]; const float* n2_b  = (const float*)d_in[18];
    const float* temp  = (const float*)d_in[19];
    const float* q_w   = (const float*)d_in[20]; const float* qd_w  = (const float*)d_in[21];
    const float* kv_w  = (const float*)d_in[22]; const float* kvd_w = (const float*)d_in[23];
    const float* po_w  = (const float*)d_in[24]; const float* pi_w  = (const float*)d_in[25];
    const float* fdw_w = (const float*)d_in[26]; const float* fpo_w = (const float*)d_in[27];

    float* outX = (float*)d_out;                        // [4,96,128,128]
    float* outL = outX + (size_t)BN * CDIM * SDIM;      // [4,1,128,128]
    float* outR = outL + (size_t)BN * 1 * SDIM;         // [4,3,128,128]

    char* ws = (char*)d_ws;
    // -------- workspace map (lifetime-safe reuse, ~212 MB total) --------
    const size_t OFF_R1  = 0;            // 50.3MB : t1/t2 -> kvf[4,192,S]f32 -> pb[4,256,S]bf16
    const size_t OFF_R2  = 50331648;     // 25.2MB : Lrein f32 -> q f32 -> xA f32
    const size_t OFF_RR  = 75497472;     // 25.2MB : Rrein f32 (until end)
    const size_t OFF_D   = 100663296;    // 12.6MB : xb bf16 -> outb bf16 -> x2b bf16
    const size_t OFF_ILL = 113246208;    // 12.6MB : illb bf16
    const size_t OFF_XN  = 125829120;    // 12.6MB : xnb bf16
    const size_t OFF_Q1  = 138412032;    // 12.6MB : q1b bf16
    const size_t OFF_KV1 = 150994944;    // 25.2MB : kv1b bf16 [4,192,S]
    const size_t OFF_FB  = 176160768;    // 33.6MB : fb bf16 [4,256,S]
    const size_t OFF_W   = 209715200;    // packed bf16 weights
    const size_t W_LC1 = OFF_W;                 // 96x864
    const size_t W_RC1 = W_LC1 + 165888;
    const size_t W_Q   = W_RC1 + 165888;        // 96x96
    const size_t W_QD  = W_Q   + 18432;         // 96x864
    const size_t W_KV  = W_QD  + 165888;        // 192x96
    const size_t W_PO  = W_KV  + 36864;         // 96x96
    const size_t W_PI  = W_PO  + 18432;         // 256x96
    const size_t W_FDW = W_PI  + 49152;         // 256x2304
    const size_t W_FPO = W_FDW + 1179648;       // 96x256
    const size_t OFF_NQ = W_FPO + 49152;        // 384 f32
    const size_t OFF_NK = OFF_NQ + 1536;
    const size_t OFF_G  = OFF_NK + 1536;        // 9216 f32
    const size_t OFF_AT = OFF_G  + 36864;       // 9216 f32

    float*  t12   = (float*)(ws + OFF_R1);
    float*  kvf   = (float*)(ws + OFF_R1);
    __bf16* pb    = (__bf16*)(ws + OFF_R1);
    float*  Lrein = (float*)(ws + OFF_R2);
    float*  q32   = (float*)(ws + OFF_R2);
    float*  xA    = (float*)(ws + OFF_R2);
    float*  Rrein = (float*)(ws + OFF_RR);
    __bf16* xb    = (__bf16*)(ws + OFF_D);
    __bf16* outb  = (__bf16*)(ws + OFF_D);
    __bf16* x2b   = (__bf16*)(ws + OFF_D);
    __bf16* illb  = (__bf16*)(ws + OFF_ILL);
    __bf16* xnb   = (__bf16*)(ws + OFF_XN);
    __bf16* q1b   = (__bf16*)(ws + OFF_Q1);
    __bf16* kv1b  = (__bf16*)(ws + OFF_KV1);
    __bf16* fb    = (__bf16*)(ws + OFF_FB);
    float*  nq    = (float*)(ws + OFF_NQ);
    float*  nk    = (float*)(ws + OFF_NK);
    float*  G     = (float*)(ws + OFF_G);
    float*  A     = (float*)(ws + OFF_AT);

    const int NR = BN * HDIM;            // 512 row-tiles (N=128 each)
    dim3 blk256g(256), blk256(256);
    #define PACK(src, dst, Co, Ci, khw, MP, KP) \
        pack_weight_k<<<((MP)*(KP)+255)/256, blk256, 0, stream>>>(src, (__bf16*)(ws+(dst)), Co, Ci, khw, MP, KP)

    PACK(Lc_w1, W_LC1,  96,  96, 9,  96,  864);
    PACK(Rc_w1, W_RC1,  96,  96, 9,  96,  864);
    PACK(q_w,   W_Q,    96,  96, 1,  96,   96);
    PACK(qd_w,  W_QD,   96,  96, 9,  96,  864);
    PACK(kv_w,  W_KV,  192,  96, 1, 192,   96);
    PACK(po_w,  W_PO,   96,  96, 1,  96,   96);
    PACK(pi_w,  W_PI,  255,  96, 1, 256,   96);
    PACK(fdw_w, W_FDW, 255, 255, 9, 256, 2304);
    PACK(fpo_w, W_FPO,  96, 255, 1,  96,  256);

    // x -> bf16
    f32_to_bf16_k<<<(BN*CDIM*SDIM+255)/256, blk256, 0, stream>>>(x, xb, BN*CDIM*SDIM);

    // ---- Retinex heads ----
    gemm_conv_bf16<9><<<dim3(NR,3), blk256g, 0, stream>>>((__bf16*)(ws+W_LC1), xb, Lc_b1,
        nullptr, nullptr, t12, nullptr, 96, 864, 96, 1);
    conv1x1_f32_k<<<(BN*1*SDIM+255)/256, blk256, 0, stream>>>(t12, Lc_w2, Lc_b2, outL, 96, 1, BN*1*SDIM);
    gemm_conv_bf16<9><<<dim3(NR,3), blk256g, 0, stream>>>((__bf16*)(ws+W_RC1), xb, Rc_b1,
        nullptr, nullptr, t12, nullptr, 96, 864, 96, 1);
    conv1x1_f32_k<<<(BN*3*SDIM+255)/256, blk256, 0, stream>>>(t12, Rc_w2, Rc_b2, outR, 96, 3, BN*3*SDIM);

    // ---- re-injection (small Cin, direct) ----
    conv3x3_direct_k<<<(BN*CDIM*SDIM+255)/256, blk256, 0, stream>>>(outL, Lr_w, Lr_b, Lrein, 1, 96, 1, BN*CDIM*SDIM);
    conv3x3_direct_k<<<(BN*CDIM*SDIM+255)/256, blk256, 0, stream>>>(outR, Rr_w, Rr_b, Rrein, 3, 96, 0, BN*CDIM*SDIM);

    // ---- LayerNorms -> bf16 ----
    layernorm_c_k<<<(BN*SDIM+255)/256, blk256, 0, stream>>>(x,     n1_w, n1_b, xnb,  96, BN*SDIM);
    layernorm_c_k<<<(BN*SDIM+255)/256, blk256, 0, stream>>>(Lrein, nL_w, nL_b, illb, 96, BN*SDIM);

    // ---- q / kv projections ----
    gemm_conv_bf16<1><<<dim3(NR,3), blk256g, 0, stream>>>((__bf16*)(ws+W_Q),  illb, nullptr,
        nullptr, nullptr, nullptr, q1b, 96, 96, 96, 0);
    gemm_conv_bf16<9><<<dim3(NR,3), blk256g, 0, stream>>>((__bf16*)(ws+W_QD), q1b,  nullptr,
        nullptr, nullptr, q32, nullptr, 96, 864, 96, 0);
    gemm_conv_bf16<1><<<dim3(NR,6), blk256g, 0, stream>>>((__bf16*)(ws+W_KV), xnb,  nullptr,
        nullptr, nullptr, nullptr, kv1b, 192, 96, 96, 0);
    dwconv3x3_k<<<(BN*192*SDIM+255)/256, blk256, 0, stream>>>(kv1b, kvd_w, kvf, 192, BN*192*SDIM);

    // ---- channel attention ----
    rownorm_k<<<BN*CDIM, blk256, 0, stream>>>(q32, nq, 96);
    rownorm_k<<<BN*CDIM, blk256, 0, stream>>>(kvf, nk, 192);
    gram_k<<<BN*NHEADS*CHD, blk256, 0, stream>>>(q32, kvf, G);
    attn_softmax_k<<<1, 384, 0, stream>>>(G, nq, nk, temp, A);
    attn_v_k<<<BN*NHEADS*(SDIM/256), blk256, 0, stream>>>(A, kvf, outb);

    // ---- project-out + residual: xA = x + conv1x1(out) ----
    gemm_conv_bf16<1><<<dim3(NR,3), blk256g, 0, stream>>>((__bf16*)(ws+W_PO), outb, nullptr,
        x, nullptr, xA, nullptr, 96, 96, 96, 0);

    // ---- FFN ----
    layernorm_c_k<<<(BN*SDIM+255)/256, blk256, 0, stream>>>(xA, n2_w, n2_b, x2b, 96, BN*SDIM);
    gemm_conv_bf16<1><<<dim3(NR,8), blk256g, 0, stream>>>((__bf16*)(ws+W_PI),  x2b, nullptr,
        nullptr, nullptr, nullptr, pb, 256, 96, 96, 0);
    gemm_conv_bf16<9><<<dim3(NR,8), blk256g, 0, stream>>>((__bf16*)(ws+W_FDW), pb,  nullptr,
        nullptr, nullptr, nullptr, fb, 256, 2304, 256, 1);
    // final: out = xA + conv1x1(gelu(...)) + R_rein   (fused epilogue)
    gemm_conv_bf16<1><<<dim3(NR,3), blk256g, 0, stream>>>((__bf16*)(ws+W_FPO), fb,  nullptr,
        xA, Rrein, outX, nullptr, 96, 256, 256, 0);
    #undef PACK
}
